// HyperbolicLinearAttention_21174188769557
// MI455X (gfx1250) — matrix operand order repaired and checked by reading, not compiled, not run on hardware
//
#include <hip/hip_runtime.h>

// ---------------------------------------------------------------------------
// HyperbolicLinearAttention on MI455X (gfx1250, wave32, WMMA)
//   kv   = K^T V  (per b,h; 64x64, reduce over N=4096)   -> phase 1
//   S    = softmax(kv, rows)                              -> phase 2
//   out  = (Q*scale) @ S^T                                -> phase 3
// bf16 WMMA (16x16x32) with f32 accumulation; deterministic reductions.
// x / out are streamed with non-temporal cache policy (read/write once,
// >> L2); the 64x64 S matrices stay temporal (reused 32x per (b,h)).
// ---------------------------------------------------------------------------

typedef __attribute__((ext_vector_type(16))) __bf16 v16bf;
typedef __attribute__((ext_vector_type(4)))  __bf16 v4bf;
typedef __attribute__((ext_vector_type(8)))  float  v8f;
typedef __attribute__((ext_vector_type(4)))  float  v4f;

constexpr int   kB   = 4;
constexpr int   kN   = 4096;
constexpr int   kH   = 32;
constexpr int   kD   = 64;
constexpr int   kC   = 3 * kH * kD;   // 6144
constexpr int   kOC  = kH * kD;       // 2048
constexpr int   kBH  = kB * kH;       // 128
constexpr int   kChunks = 8;          // N split for phase-1 parallelism
constexpr int   kChunkN = kN / kChunks; // 512
constexpr int   kTN  = 64;            // phase-1 n-tile per iteration
constexpr float kScale = 0.17677669529663687f; // 32^-0.5

constexpr int LROW  = kTN + 4;        // 68: phase-1 LDS row stride (n-dim)
constexpr int SROW  = kD + 8;         // 72: phase-3 LDS row stride (d-dim)

__device__ __forceinline__ __bf16 f2bf(float f) { return (__bf16)f; }

// ---------------------------------------------------------------------------
// Phase 1: partial kv[d][e] = sum_n K[n][d] * V[n][e] over one N-chunk.
// Block = 256 thr (8 waves). Wave w owns output tiles (d0=(w/2)*16,
// e0=(w&1)*32 .. +16). K/V staged transposed in LDS as bf16: each thread
// owns 4 consecutive n of 4 d-columns -> packed v4bf (ds_store_b64) writes.
// ---------------------------------------------------------------------------
__global__ void __launch_bounds__(256)
hla_kv_partial(const float* __restrict__ x, float* __restrict__ kv_part) {
  __shared__ __bf16 lds_kT[kD * LROW];   // [d][n]  (transposed)
  __shared__ __bf16 lds_vT[kD * LROW];   // [e][n]  (transposed)

  const int bh    = blockIdx.x;
  const int b     = bh / kH, h = bh % kH;
  const int chunk = blockIdx.y;
  const int tid   = threadIdx.x;
  const int wave  = tid >> 5;
  const int lane  = tid & 31;
  const int half  = lane >> 4;
  const int m     = lane & 15;

  const int d0 = (wave >> 1) * 16;
  const int e0 = (wave & 1) * 32;

  // loader assignment: 16 d-column groups x 16 row groups of 4
  const int col4    = tid & 15;         // d/4
  const int rowbase = (tid >> 4) * 4;   // n-local 0..60

  const float* kbase = x + (size_t)b * kN * kC + kOC     + h * kD;
  const float* vbase = x + (size_t)b * kN * kC + 2 * kOC + h * kD;

  v8f c0 = {}; v8f c1 = {};

  for (int it = 0; it < kChunkN / kTN; ++it) {
    const int nbase = chunk * kChunkN + it * kTN;
    __syncthreads();

    // ---- cooperative transposed load: 4 rows x 4 cols per thread ----
    v4f kf[4], vf[4];
#pragma unroll
    for (int j = 0; j < 4; ++j) {
      const size_t roff = (size_t)(nbase + rowbase + j) * kC + col4 * 4;
      kf[j] = __builtin_nontemporal_load((const v4f*)(kbase + roff));
      vf[j] = __builtin_nontemporal_load((const v4f*)(vbase + roff));
    }
#pragma unroll
    for (int c = 0; c < 4; ++c) {
      v4bf kp, vp;
#pragma unroll
      for (int j = 0; j < 4; ++j) {
        kp[j] = f2bf(kf[j][c]);
        vp[j] = f2bf(vf[j][c]);
      }
      const int d = col4 * 4 + c;
      *(v4bf*)&lds_kT[d * LROW + rowbase] = kp;   // ds_store_b64
      *(v4bf*)&lds_vT[d * LROW + rowbase] = vp;
    }
    __syncthreads();

    // prefetch next n-tile while this one computes (global_prefetch_b8)
    if (it + 1 < kChunkN / kTN) {
      const int pr = tid & 63;
      if (tid < 64)        __builtin_prefetch(kbase + (size_t)(nbase + kTN + pr) * kC, 0, 1);
      else if (tid < 128)  __builtin_prefetch(vbase + (size_t)(nbase + kTN + pr) * kC, 0, 1);
    }

    // ---- 2 K-steps of 32 over the 64-row tile ----
#pragma unroll
    for (int ks = 0; ks < 2; ++ks) {
      const int nb = ks * 32;
      // A = K^T tile 16(M=d) x 32(K=n): per-lane K runs {h*8.., 16+h*8..}
      v16bf a, b0, b1;
#pragma unroll
      for (int i = 0; i < 8; ++i) {
        a[i]     = lds_kT[(d0 + m) * LROW + nb + half * 8 + i];
        a[8 + i] = lds_kT[(d0 + m) * LROW + nb + 16 + half * 8 + i];
      }
      // B = V tile 32(K=n) x 16(N=e): per-lane K runs {h*8.., 16+h*8..}
#pragma unroll
      for (int i = 0; i < 8; ++i) {
        b0[i]     = lds_vT[(e0 + m)      * LROW + nb + half * 8 + i];
        b0[8 + i] = lds_vT[(e0 + m)      * LROW + nb + 16 + half * 8 + i];
        b1[i]     = lds_vT[(e0 + 16 + m) * LROW + nb + half * 8 + i];
        b1[8 + i] = lds_vT[(e0 + 16 + m) * LROW + nb + 16 + half * 8 + i];
      }
      c0 = __builtin_amdgcn_wmma_f32_16x16x32_bf16(false, a, false, b0, (short)0, c0, false, false);
      c1 = __builtin_amdgcn_wmma_f32_16x16x32_bf16(false, a, false, b1, (short)0, c1, false, false);
    }
  }

  // deterministic partial write-out: one slab per chunk (streamed, NT)
  float* dst = kv_part + ((size_t)chunk * kBH + bh) * (kD * kD);
#pragma unroll
  for (int r = 0; r < 8; ++r) {
    const int drow = d0 + r + 8 * half;
    __builtin_nontemporal_store(c0[r], &dst[drow * kD + e0 + m]);
    __builtin_nontemporal_store(c1[r], &dst[drow * kD + e0 + 16 + m]);
  }
}

// ---------------------------------------------------------------------------
// Phase 2: sum the 8 partials (deterministic order), row softmax, store bf16
// S[e][d] row-major (exactly the B-operand layout phase 3 wants).
// One wave per 64-element row; 8 rows per 256-thread block.
// ---------------------------------------------------------------------------
__global__ void __launch_bounds__(256)
hla_softmax(const float* __restrict__ kv_part, __bf16* __restrict__ kv_soft) {
  const int tid  = threadIdx.x;
  const int wave = tid >> 5, lane = tid & 31;
  const int row  = blockIdx.x * 8 + wave;   // 0 .. kBH*64-1
  const int bh   = row >> 6, r = row & 63;

  const size_t matoff = (size_t)bh * (kD * kD) + (size_t)r * kD;
  float v0 = 0.f, v1 = 0.f;
#pragma unroll
  for (int ch = 0; ch < kChunks; ++ch) {
    const float* p = kv_part + (size_t)ch * kBH * (kD * kD) + matoff;
    v0 += __builtin_nontemporal_load(&p[lane]);
    v1 += __builtin_nontemporal_load(&p[lane + 32]);
  }
  float mx = fmaxf(v0, v1);
#pragma unroll
  for (int off = 16; off > 0; off >>= 1) mx = fmaxf(mx, __shfl_xor(mx, off, 32));
  const float e0 = __expf(v0 - mx);
  const float e1 = __expf(v1 - mx);
  float s = e0 + e1;
#pragma unroll
  for (int off = 16; off > 0; off >>= 1) s += __shfl_xor(s, off, 32);
  const float inv = 1.0f / s;

  __bf16* dst = kv_soft + matoff;   // temporal: reused 32x by phase 3
  dst[lane]      = f2bf(e0 * inv);
  dst[lane + 32] = f2bf(e1 * inv);
}

// ---------------------------------------------------------------------------
// Phase 3: out[n][e] = sum_d (Q[n][d]*scale) * S[e][d].
// Block = 256 thr, handles a 128-row n-strip of one (b,h); S staged once in
// LDS (9 KB). Wave computes a 16x64 strip: 4 C tiles x 2 K-steps = 8 WMMAs.
// Q loaded as 16-B NT vectors (A-layout K-runs of 8 are contiguous).
// ---------------------------------------------------------------------------
__global__ void __launch_bounds__(256)
hla_attn(const float* __restrict__ x, const __bf16* __restrict__ kv_soft,
         float* __restrict__ out) {
  __shared__ __bf16 lds_s[kD * SROW];   // S[e][d], padded rows

  const int bh   = blockIdx.x;
  const int b    = bh / kH, h = bh % kH;
  const int tid  = threadIdx.x;
  const int wave = tid >> 5, lane = tid & 31;
  const int half = lane >> 4, m = lane & 15;

  // stage S (64x64 bf16, coalesced, temporal)
#pragma unroll
  for (int j = 0; j < 16; ++j) {
    const int idx = j * 256 + tid;
    lds_s[(idx >> 6) * SROW + (idx & 63)] = kv_soft[(size_t)bh * (kD * kD) + idx];
  }
  __syncthreads();

  const int n0 = blockIdx.y * 128 + wave * 16;
  const float* q = x + ((size_t)b * kN + n0 + m) * kC + h * kD;

  // A = Q tile 16(M=n) x 32(K=d), two K-steps (d 0..31, 32..63), scaled bf16
  const v4f q00 = __builtin_nontemporal_load((const v4f*)(q + half * 8));
  const v4f q01 = __builtin_nontemporal_load((const v4f*)(q + half * 8 + 4));
  const v4f q02 = __builtin_nontemporal_load((const v4f*)(q + 16 + half * 8));
  const v4f q03 = __builtin_nontemporal_load((const v4f*)(q + 20 + half * 8));
  const v4f q10 = __builtin_nontemporal_load((const v4f*)(q + 32 + half * 8));
  const v4f q11 = __builtin_nontemporal_load((const v4f*)(q + 36 + half * 8));
  const v4f q12 = __builtin_nontemporal_load((const v4f*)(q + 48 + half * 8));
  const v4f q13 = __builtin_nontemporal_load((const v4f*)(q + 52 + half * 8));

  v16bf a0, a1;
#pragma unroll
  for (int i = 0; i < 4; ++i) {
    a0[i]      = f2bf(q00[i] * kScale);
    a0[4 + i]  = f2bf(q01[i] * kScale);
    a0[8 + i]  = f2bf(q02[i] * kScale);
    a0[12 + i] = f2bf(q03[i] * kScale);
    a1[i]      = f2bf(q10[i] * kScale);
    a1[4 + i]  = f2bf(q11[i] * kScale);
    a1[8 + i]  = f2bf(q12[i] * kScale);
    a1[12 + i] = f2bf(q13[i] * kScale);
  }

  float* obase = out + ((size_t)b * kN + n0) * kOC + h * kD;
#pragma unroll
  for (int et = 0; et < 4; ++et) {
    // B[k=d][col=e] = S[e][d]: lane col e = et*16+m, contiguous d runs in LDS
    v16bf b0, b1;
#pragma unroll
    for (int i = 0; i < 16; ++i) {
      b0[i] = lds_s[(et * 16 + m) * SROW + (i >> 3) * 16 + half * 8 + (i & 7)];        // d = 0..31
      b1[i] = lds_s[(et * 16 + m) * SROW + 32 + (i >> 3) * 16 + half * 8 + (i & 7)];   // d = 32..63
    }
    v8f c = {};
    c = __builtin_amdgcn_wmma_f32_16x16x32_bf16(false, a0, false, b0, (short)0, c, false, false);
    c = __builtin_amdgcn_wmma_f32_16x16x32_bf16(false, a1, false, b1, (short)0, c, false, false);
#pragma unroll
    for (int r = 0; r < 8; ++r) {
      __builtin_nontemporal_store(c[r],
          &obase[(size_t)(r + 8 * half) * kOC + et * 16 + m]);
    }
  }
}

// ---------------------------------------------------------------------------
extern "C" void kernel_launch(void* const* d_in, const int* in_sizes, int n_in,
                              void* d_out, int out_size, void* d_ws, size_t ws_size,
                              hipStream_t stream) {
  (void)in_sizes; (void)n_in; (void)out_size; (void)ws_size;
  const float* x = (const float*)d_in[0];
  float* out = (float*)d_out;

  // workspace: [0, 16MB) fp32 kv partials (8 chunks x 128 mats x 64x64),
  //            [16MB, 17MB) bf16 softmaxed S matrices
  float*  kv_part = (float*)d_ws;
  __bf16* kv_soft = (__bf16*)((char*)d_ws +
                    (size_t)kChunks * kBH * kD * kD * sizeof(float));

  hla_kv_partial<<<dim3(kBH, kChunks), 256, 0, stream>>>(x, kv_part);
  hla_softmax<<<(kBH * kD) / 8, 256, 0, stream>>>(kv_part, kv_soft);
  hla_attn<<<dim3(kBH, kN / 128), 256, 0, stream>>>(x, kv_soft, out);
}
